// GraphAttnBias_48773648614060
// MI455X (gfx1250) — compile-verified, matrix operand order checked by reading, not run on hardware
//
#include <hip/hip_runtime.h>

typedef __attribute__((ext_vector_type(2)))  float    v2f;
typedef __attribute__((ext_vector_type(4)))  float    v4f;
typedef __attribute__((ext_vector_type(8)))  float    v8f;
typedef __attribute__((ext_vector_type(8)))  _Float16 v8h;
typedef __attribute__((ext_vector_type(16))) _Float16 v16h;

#define BB 8
#define NN 512
#define SS 64
#define HEADS 16
#define DD 256

// LDS byte-offset of a generic pointer into __shared__ (low 32 bits of flat VA)
__device__ __forceinline__ unsigned lds_off(const void* p) {
  return (unsigned)(unsigned long long)p;
}

// Async global->LDS 16-byte copy (CDNA5 path, tracked by ASYNCcnt)
__device__ __forceinline__ void async_b128(unsigned lds_byte, const void* gaddr) {
  asm volatile("global_load_async_to_lds_b128 %0, %1, off"
               :: "v"(lds_byte), "v"((unsigned long long)gaddr)
               : "memory");
}
__device__ __forceinline__ void wait_async0() {
  asm volatile("s_wait_asynccnt 0x0" ::: "memory");
}

// ---------------- K1: per-row deg^-0.5 and h2 -------------------------------
__global__ void __launch_bounds__(256) k_prep_rows(const float* __restrict__ A,
                                                   const float* __restrict__ H,
                                                   float* __restrict__ rsdeg,
                                                   float* __restrict__ h2) {
  int wave = threadIdx.x >> 5, lane = threadIdx.x & 31;
  int row = blockIdx.x * 8 + wave;              // row = b*N + n, 0..4095
  const float* arow = A + (size_t)row * NN;
  float s = 0.f;
  for (int i = lane; i < NN; i += 32) s += arow[i];
  #pragma unroll
  for (int off = 16; off; off >>= 1) s += __shfl_xor(s, off, 32);
  const float* hrow = H + (size_t)row * DD;
  float s2 = 0.f;
  for (int i = lane; i < DD; i += 32) { float v = hrow[i]; s2 += v * v; }
  #pragma unroll
  for (int off = 16; off; off >>= 1) s2 += __shfl_xor(s2, off, 32);
  if (lane == 0) { rsdeg[row] = rsqrtf(s); h2[row] = s2; }
}

// ---------------- K2: gather embeddings -> f16 V[b,h,n,s] -------------------
// Handles the contiguous .view(b,heads,n,s) reinterpretation of [b,n,s,heads].
__global__ void __launch_bounds__(256) k_gather_emb(const int* __restrict__ pos,
                                                    const float* __restrict__ emb,
                                                    _Float16* __restrict__ V) {
  int idx = blockIdx.x * 256 + threadIdx.x;     // < B*HEADS*N*S
  int s = idx & 63;
  int n = (idx >> 6) & 511;
  int h = (idx >> 15) & 15;
  int b = idx >> 19;
  int lin = h * (NN * SS) + n * SS + s;         // flat index in [heads,N,S] view
  int n0 = lin >> 10;                           // original n   ( /(S*HEADS) )
  int s0 = (lin >> 4) & 63;                     // original s
  int h0 = lin & 15;                            // original head
  int p = pos[(b * NN + n0) * SS + s0];
  float v = (p == 0) ? 0.f : emb[p * HEADS + h0];   // padding_idx = 0
  V[idx] = (_Float16)v;
}

// ---------------- K3: M = (deg^-1/2 * A) @ H, f32 WMMA ----------------------
__global__ void __launch_bounds__(256) k_norA_H(const float* __restrict__ A,
                                                const float* __restrict__ H,
                                                const float* __restrict__ rsdeg,
                                                float* __restrict__ M) {
  __shared__ float sA[16][516];                 // 16 scaled A rows, padded stride
  int b = blockIdx.z, nt = blockIdx.y, dchunk = blockIdx.x;
  int t = threadIdx.x;
  {
    int r = t >> 4;                             // 0..15
    int c0 = (t & 15) * 32;
    int grow = b * NN + nt * 16 + r;
    float rs = rsdeg[grow];
    const float* arow = A + (size_t)grow * NN + c0;
    #pragma unroll
    for (int q = 0; q < 32; q += 4) {
      v4f v = *(const v4f*)(arow + q);
      v = v * rs;
      *(v4f*)&sA[r][c0 + q] = v;
    }
  }
  __syncthreads();
  int wave = t >> 5, lane = t & 31, L2 = lane & 15, hi = lane >> 4;
  int dcol = dchunk * 128 + wave * 16 + L2;
  v8f acc = {};
  for (int k = 0; k < NN; k += 4) {
    v2f a = *(const v2f*)&sA[L2][k + 2 * hi];
    v2f bf;
    bf[0] = H[(size_t)(b * NN + k + 2 * hi)     * DD + dcol];
    bf[1] = H[(size_t)(b * NN + k + 2 * hi + 1) * DD + dcol];
    acc = __builtin_amdgcn_wmma_f32_16x16x4_f32(false, a, false, bf,
                                                (short)0, acc, false, false);
  }
  int nbase = b * NN + nt * 16 + hi * 8;
  #pragma unroll
  for (int r = 0; r < 8; r++)
    M[(size_t)(nbase + r) * DD + dcol] = acc[r];
}

// ---------------- K4: m2 = rowsum(M^2) --------------------------------------
__global__ void __launch_bounds__(256) k_rowsumsq(const float* __restrict__ M,
                                                  float* __restrict__ m2) {
  int wave = threadIdx.x >> 5, lane = threadIdx.x & 31;
  int row = blockIdx.x * 8 + wave;
  const float* mr = M + (size_t)row * DD;
  float s = 0.f;
  for (int i = lane; i < DD; i += 32) { float v = mr[i]; s += v * v; }
  #pragma unroll
  for (int off = 16; off; off >>= 1) s += __shfl_xor(s, off, 32);
  if (lane == 0) m2[row] = s;
}

// ---------------- K5 helper: one 16x16 bias subtile (f16 WMMA, K=64) --------
__device__ __forceinline__ v8f sb_tile(const _Float16* Vn, const _Float16* Vm,
                                       int ti, int tj, int L2, int hi, v8f c) {
  #pragma unroll
  for (int k0 = 0; k0 < 64; k0 += 32) {
    const _Float16* ar = Vn + (ti * 16 + L2) * 72 + k0 + hi * 8;
    const _Float16* br = Vm + (tj * 16 + L2) * 72 + k0 + hi * 8;
    v8h a0 = *(const v8h*)ar;
    v8h a1 = *(const v8h*)(ar + 16);
    v8h b0 = *(const v8h*)br;
    v8h b1 = *(const v8h*)(br + 16);
    v16h av = __builtin_shufflevector(a0, a1, 0,1,2,3,4,5,6,7,8,9,10,11,12,13,14,15);
    v16h bv = __builtin_shufflevector(b0, b1, 0,1,2,3,4,5,6,7,8,9,10,11,12,13,14,15);
    c = __builtin_amdgcn_wmma_f32_16x16x32_f16(false, av, false, bv,
                                               (short)0, c, false, false);
  }
  return c;
}

// ---------------- K5: fused 64x64 output tile, all 16 heads -----------------
__global__ void __launch_bounds__(256) k_fused_out(
    const float* __restrict__ H, const float* __restrict__ M,
    const float* __restrict__ h2, const float* __restrict__ m2,
    const _Float16* __restrict__ V, const float* __restrict__ sigma,
    float* __restrict__ out) {
  __shared__ float sg[64][65];                              // gaussian tile
  __shared__ __align__(16) unsigned char sbuf[2 * 64 * 68 * 4]; // H/M then V tiles
  float*    Ht = (float*)sbuf;                              // [64][68]
  float*    Mt = (float*)sbuf + 64 * 68;                    // [64][68]
  _Float16* Vn = (_Float16*)sbuf;                           // [64][72]
  _Float16* Vm = (_Float16*)sbuf + 64 * 72;                 // [64][72]
  unsigned htB = lds_off(Ht), mtB = lds_off(Mt);
  unsigned vnB = lds_off(Vn), vmB = lds_off(Vm);

  int b = blockIdx.z, n0 = blockIdx.y * 64, m0 = blockIdx.x * 64;
  int t = threadIdx.x;
  int wave = t >> 5, lane = t & 31, L2 = lane & 15, hi = lane >> 4;
  int ti0 = (wave * 2) >> 2,     tj0 = (wave * 2) & 3;
  int ti1 = (wave * 2 + 1) >> 2, tj1 = (wave * 2 + 1) & 3;
  int lr = t >> 2;                 // cooperative-load row 0..63
  int lc = (t & 3) * 16;           // cooperative-load col base

  // ---- Phase 1: cross = H @ M^T via f32 WMMA, K chunked through LDS ----
  v8f acc0 = {}, acc1 = {};
  for (int kc = 0; kc < DD; kc += 64) {
    const float* hsrc = H + (size_t)(b * NN + n0 + lr) * DD + kc + lc;
    const float* msrc = M + (size_t)(b * NN + m0 + lr) * DD + kc + lc;
    #pragma unroll
    for (int q = 0; q < 16; q += 4) {
      async_b128(htB + (unsigned)(lr * 68 + lc + q) * 4, hsrc + q);
      async_b128(mtB + (unsigned)(lr * 68 + lc + q) * 4, msrc + q);
    }
    wait_async0();
    __syncthreads();
    for (int k = 0; k < 64; k += 4) {
      v2f a0 = *(const v2f*)&Ht[(ti0 * 16 + L2) * 68 + k + 2 * hi];
      v2f b0 = *(const v2f*)&Mt[(tj0 * 16 + L2) * 68 + k + 2 * hi];
      acc0 = __builtin_amdgcn_wmma_f32_16x16x4_f32(false, a0, false, b0,
                                                   (short)0, acc0, false, false);
      v2f a1 = *(const v2f*)&Ht[(ti1 * 16 + L2) * 68 + k + 2 * hi];
      v2f b1 = *(const v2f*)&Mt[(tj1 * 16 + L2) * 68 + k + 2 * hi];
      acc1 = __builtin_amdgcn_wmma_f32_16x16x4_f32(false, a1, false, b1,
                                                   (short)0, acc1, false, false);
    }
    __syncthreads();
  }

  // ---- gaussian tile into LDS ----
  float sval = sigma[0];
  float inv2s2 = 1.0f / (2.0f * sval * sval);
  {
    float m2v = m2[b * NN + m0 + tj0 * 16 + L2];
    #pragma unroll
    for (int r = 0; r < 8; r++) {
      int row = ti0 * 16 + hi * 8 + r;
      float d2 = h2[b * NN + n0 + row] + m2v - 2.0f * acc0[r];
      sg[row][tj0 * 16 + L2] = __expf(-d2 * inv2s2);
    }
  }
  {
    float m2v = m2[b * NN + m0 + tj1 * 16 + L2];
    #pragma unroll
    for (int r = 0; r < 8; r++) {
      int row = ti1 * 16 + hi * 8 + r;
      float d2 = h2[b * NN + n0 + row] + m2v - 2.0f * acc1[r];
      sg[row][tj1 * 16 + L2] = __expf(-d2 * inv2s2);
    }
  }

  // ---- Phase 2: per-head spatial bias (f16 WMMA), C-init = gaussian ----
  for (int h = 0; h < HEADS; h++) {
    __syncthreads();   // prior readers of sbuf done (also fences sg writes)
    const _Float16* vn = V + (size_t)((b * HEADS + h) * NN + n0 + lr) * SS + lc;
    const _Float16* vm = V + (size_t)((b * HEADS + h) * NN + m0 + lr) * SS + lc;
    async_b128(vnB + (unsigned)(lr * 72 + lc) * 2,       vn);
    async_b128(vnB + (unsigned)(lr * 72 + lc + 8) * 2,   vn + 8);
    async_b128(vmB + (unsigned)(lr * 72 + lc) * 2,       vm);
    async_b128(vmB + (unsigned)(lr * 72 + lc + 8) * 2,   vm + 8);
    if (h + 1 < HEADS) {                         // warm L2 for next head
      __builtin_prefetch(vn + (size_t)NN * SS, 0, 1);
      __builtin_prefetch(vm + (size_t)NN * SS, 0, 1);
    }
    wait_async0();
    __syncthreads();

    float* op = out + (size_t)((b * HEADS + h) * NN + n0) * NN + m0;
    {
      v8f c;
      #pragma unroll
      for (int r = 0; r < 8; r++) c[r] = sg[ti0 * 16 + hi * 8 + r][tj0 * 16 + L2];
      c = sb_tile(Vn, Vm, ti0, tj0, L2, hi, c);
      #pragma unroll
      for (int r = 0; r < 8; r++)
        op[(size_t)(ti0 * 16 + hi * 8 + r) * NN + tj0 * 16 + L2] = c[r];
    }
    {
      v8f c;
      #pragma unroll
      for (int r = 0; r < 8; r++) c[r] = sg[ti1 * 16 + hi * 8 + r][tj1 * 16 + L2];
      c = sb_tile(Vn, Vm, ti1, tj1, L2, hi, c);
      #pragma unroll
      for (int r = 0; r < 8; r++)
        op[(size_t)(ti1 * 16 + hi * 8 + r) * NN + tj1 * 16 + L2] = c[r];
    }
  }
}

// ---------------------------------------------------------------------------
extern "C" void kernel_launch(void* const* d_in, const int* in_sizes, int n_in,
                              void* d_out, int out_size, void* d_ws, size_t ws_size,
                              hipStream_t stream) {
  (void)in_sizes; (void)n_in; (void)out_size; (void)ws_size;
  const int*   pos   = (const int*)d_in[0];
  const float* H     = (const float*)d_in[1];
  const float* A     = (const float*)d_in[2];
  const float* sigma = (const float*)d_in[3];
  const float* emb   = (const float*)d_in[4];
  // d_in[5] (vt_weight) provably has no effect on the returned slice.
  float* out = (float*)d_out;

  char* ws = (char*)d_ws;
  float*    M     = (float*)ws;                                   // 4 MB
  _Float16* V     = (_Float16*)(ws + ((size_t)4 << 20));          // 8 MB
  float*    rsdeg = (float*)(ws + ((size_t)12 << 20));            // 16 KB
  float*    h2    = rsdeg + BB * NN;
  float*    m2    = h2 + BB * NN;

  k_prep_rows <<<BB * NN / 8, 256, 0, stream>>>(A, H, rsdeg, h2);
  k_gather_emb<<<(BB * HEADS * NN * SS) / 256, 256, 0, stream>>>(pos, emb, V);
  k_norA_H    <<<dim3(2, 32, 8), 256, 0, stream>>>(A, H, rsdeg, M);
  k_rowsumsq  <<<BB * NN / 8, 256, 0, stream>>>(M, m2);
  k_fused_out <<<dim3(8, 8, 8), 256, 0, stream>>>(H, M, h2, m2, V, sigma, out);
}
